// MSARowAttentionWithPairBias_43044162241201
// MI455X (gfx1250) — matrix operand order repaired and checked by reading, not compiled, not run on hardware
//
#include <hip/hip_runtime.h>
#include <hip/hip_bf16.h>
#include <stdint.h>
#include <stddef.h>

// ---------------------------------------------------------------------------
// Problem constants (fixed by the reference)
// ---------------------------------------------------------------------------
// B=64, S=512, c_m=256, c_z=128, H=8, D=32
constexpr int kB  = 64;
constexpr int kS  = 512;
constexpr int kCM = 256;
constexpr int kCZ = 128;
constexpr int kH  = 8;
constexpr int kD  = 32;

typedef __attribute__((ext_vector_type(16))) __bf16        v16bf;
typedef __attribute__((ext_vector_type(8)))  float         v8f;
typedef __attribute__((ext_vector_type(4)))  unsigned int  v4u;

union V16 { v16bf b; v4u u[2]; };

// A operand as loaded here (contiguous k order, matched by B): lane holds row (lane&15);
// vector elems [0..15] = K kbase+0..15, where kbase = (lane>>4)*16.
// 32 contiguous bytes in row-major memory -> two b128 loads.
__device__ inline v16bf loadA(const __bf16* p) {
  V16 x;
  x.u[0] = *(const v4u*)(p);
  x.u[1] = *(const v4u*)(p + 8);
  return x.b;
}

// B operand as loaded here (same contiguous k order as A): lane holds col (lane&15); per-lane
// 16 contiguous K values starting at (lane>>4)*16 -> 32 contiguous bytes.
__device__ inline v16bf loadB(const __bf16* p) {
  V16 x;
  x.u[0] = *(const v4u*)(p);
  x.u[1] = *(const v4u*)(p + 8);
  return x.b;
}

__device__ inline float wave_sum32(float v) {
  #pragma unroll
  for (int m = 16; m >= 1; m >>= 1) v += __shfl_xor(v, m, 32);
  return v;
}
// Reductions across a 16-lane half (N dimension of a C/D tile).
__device__ inline float half_max16(float v) {
  #pragma unroll
  for (int m = 8; m >= 1; m >>= 1) v = fmaxf(v, __shfl_xor(v, m, 32));
  return v;
}
__device__ inline float half_sum16(float v) {
  #pragma unroll
  for (int m = 8; m >= 1; m >>= 1) v += __shfl_xor(v, m, 32);
  return v;
}

// ---------------------------------------------------------------------------
// 1. Weight prep: f32 W[k][n] -> bf16 W^T[n][k]
// ---------------------------------------------------------------------------
__global__ void prep_w_kernel(const float* __restrict__ W, __bf16* __restrict__ WT) {
  int i = blockIdx.x * blockDim.x + threadIdx.x;       // over 256*256
  int kk = i >> 8;
  int n  = i & 255;
  WT[n * 256 + kk] = (__bf16)W[i];
}

// ---------------------------------------------------------------------------
// 2. RMSNorm(m) -> bf16 rows (wave per row, 8 elems/lane)
// ---------------------------------------------------------------------------
__global__ void rmsnorm_m_kernel(const float* __restrict__ m,
                                 const float* __restrict__ w,
                                 __bf16* __restrict__ out) {
  int wid  = (blockIdx.x * blockDim.x + threadIdx.x) >> 5;   // row id
  int lane = threadIdx.x & 31;
  const float* row = m + (size_t)wid * kCM;
  float v[8];
  float ss = 0.f;
  #pragma unroll
  for (int j = 0; j < 8; ++j) { v[j] = row[lane * 8 + j]; ss += v[j] * v[j]; }
  ss = wave_sum32(ss);
  float r = rsqrtf(ss * (1.0f / kCM) + 1e-5f);
  __bf16* orow = out + (size_t)wid * kCM;
  #pragma unroll
  for (int j = 0; j < 8; ++j)
    orow[lane * 8 + j] = (__bf16)(v[j] * r * w[lane * 8 + j]);
}

// ---------------------------------------------------------------------------
// 3. RMSNorm(z) + pair bias + mask, fused single pass over z.
//    One wave per (i,j) pair; bias[h][i][j] = z_norm(i,j,:) . Wz[:,h] + maskbias
// ---------------------------------------------------------------------------
__global__ void pairbias_kernel(const float* __restrict__ z,
                                const float* __restrict__ zmask,
                                const float* __restrict__ wnz,
                                const float* __restrict__ Wz,
                                float* __restrict__ bias) {
  int wid  = (blockIdx.x * blockDim.x + threadIdx.x) >> 5;   // (i*512 + j)
  int lane = threadIdx.x & 31;
  const float* zr = z + (size_t)wid * kCZ;
  float v[4];
  float ss = 0.f;
  #pragma unroll
  for (int t = 0; t < 4; ++t) { v[t] = zr[lane * 4 + t]; ss += v[t] * v[t]; }
  ss = wave_sum32(ss);
  float r = rsqrtf(ss * (1.0f / kCZ) + 1e-5f);
  #pragma unroll
  for (int t = 0; t < 4; ++t) v[t] = v[t] * r * wnz[lane * 4 + t];
  float mb = (zmask[wid] > 0.f) ? 0.f : -1.0e9f;
  #pragma unroll
  for (int h = 0; h < kH; ++h) {
    float d = 0.f;
    #pragma unroll
    for (int t = 0; t < 4; ++t) d += v[t] * Wz[(lane * 4 + t) * kH + h];
    d = wave_sum32(d);
    if (lane == h) bias[((size_t)h << 18) + wid] = d + mb;
  }
}

// ---------------------------------------------------------------------------
// 4. Generic wave-tiled WMMA GEMM: Y[M x 256] = A[M x 256] * B(=BT^T)[256 x 256]
//    One wave computes one 16x16 tile; K swept in 8 chunks of 32.
//    mode 0: bf16 row-major out, val*scale          (Q / K / attn-A stores)
//    mode 2: bf16 per-head transposed out (V^T)
//    mode 3: f32 out = val + bvec[col]              (G)
//    mode 4: f32 out = (val + bvec[col])*gmat       (final O, fused gate)
// ---------------------------------------------------------------------------
__global__ void gemm16_kernel(const __bf16* __restrict__ A,
                              const __bf16* __restrict__ BT,
                              float scale,
                              const float* __restrict__ bvec,
                              const float* __restrict__ gmat,
                              __bf16* __restrict__ obf,
                              float*  __restrict__ ofp,
                              int mode) {
  const int lane  = threadIdx.x & 31;
  const int wid   = (blockIdx.x * blockDim.x + threadIdx.x) >> 5;
  const int mtile = wid >> 4;          // 16 ntiles (N = 256)
  const int ntile = wid & 15;
  const int ln    = lane & 15;
  const int lg    = lane >> 4;

  const __bf16* arow = A  + (size_t)(mtile * 16 + ln) * kCM + lg * 16;
  const __bf16* brow = BT + (size_t)(ntile * 16 + ln) * kCM + lg * 16;

  v8f acc = {};
  #pragma unroll
  for (int kc = 0; kc < 8; ++kc) {
    v16bf a = loadA(arow + kc * 32);
    v16bf b = loadB(brow + kc * 32);
    acc = __builtin_amdgcn_wmma_f32_16x16x32_bf16(false, a, false, b,
                                                  (short)0, acc, false, false);
  }

  const int mrow0 = mtile * 16 + lg * 8;   // rows mrow0..mrow0+7 in this half
  const int col   = ntile * 16 + ln;
  #pragma unroll
  for (int r = 0; r < 8; ++r) {
    int row   = mrow0 + r;
    float val = acc[r];
    if (mode == 0) {
      obf[(size_t)row * kCM + col] = (__bf16)(val * scale);
    } else if (mode == 2) {
      int b_ = row >> 9, s_ = row & 511;
      int hh = col >> 5, d  = col & 31;
      obf[(((size_t)((b_ * kH + hh) * kD + d)) << 9) + s_] = (__bf16)val;
    } else if (mode == 3) {
      ofp[(size_t)row * kCM + col] = val + bvec[col];
    } else {
      ofp[(size_t)row * kCM + col] =
          (val + bvec[col]) * gmat[(size_t)row * kCM + col];
    }
  }
}

// ---------------------------------------------------------------------------
// 5. Flash attention: one wave per (b, h, 16-query tile).
//    Per 32-key chunk: QK^T (2 WMMA), +bias, online softmax, P through LDS
//    (D-layout -> A-layout transpose), P*V (2 WMMA).
// ---------------------------------------------------------------------------
__global__ void __launch_bounds__(32)
attn_kernel(const __bf16* __restrict__ q, const __bf16* __restrict__ k,
            const __bf16* __restrict__ vT, const float* __restrict__ bias,
            __bf16* __restrict__ aout) {
  __shared__ __align__(16) __bf16 p_lds[16 * 32];

  const int lane  = threadIdx.x;
  const int wid   = blockIdx.x;
  const int qt    = wid & 31;          // S/16 query tiles
  const int h     = (wid >> 5) & 7;
  const int b     = wid >> 8;
  const int qbase = qt * 16;
  const int ln    = lane & 15;
  const int lg    = lane >> 4;

  // Q tile as A operand: K = full head dim (32)
  v16bf qa = loadA(q + ((size_t)(b * kS + qbase + ln) * kCM) + h * kD + lg * 16);

  v8f acc0 = {}, acc1 = {};
  float mrow[8], lrow[8];
  #pragma unroll
  for (int r = 0; r < 8; ++r) { mrow[r] = -3.0e38f; lrow[r] = 0.f; }

  const v8f zero = {};
  for (int c = 0; c < kS / 32; ++c) {
    const int base = c * 32;

    // ---- scores: s = (q*scale) @ k^T, two 16-key tiles -------------------
    v16bf kb0 = loadB(k + ((size_t)(b * kS + base + ln)      * kCM) + h * kD + lg * 16);
    v16bf kb1 = loadB(k + ((size_t)(b * kS + base + 16 + ln) * kCM) + h * kD + lg * 16);
    v8f s0 = __builtin_amdgcn_wmma_f32_16x16x32_bf16(false, qa, false, kb0,
                                                     (short)0, zero, false, false);
    v8f s1 = __builtin_amdgcn_wmma_f32_16x16x32_bf16(false, qa, false, kb1,
                                                     (short)0, zero, false, false);

    // ---- + pair bias (incl. mask) ---------------------------------------
    #pragma unroll
    for (int r = 0; r < 8; ++r) {
      size_t boff = ((size_t)(h * kS + qbase + lg * 8 + r)) * kS + base + ln;
      s0[r] += bias[boff];
      s1[r] += bias[boff + 16];
    }

    // ---- online softmax --------------------------------------------------
    #pragma unroll
    for (int r = 0; r < 8; ++r) {
      float cmx = half_max16(fmaxf(s0[r], s1[r]));
      float mn  = fmaxf(mrow[r], cmx);
      float alpha = __expf(mrow[r] - mn);
      mrow[r] = mn;
      s0[r] = __expf(s0[r] - mn);
      s1[r] = __expf(s1[r] - mn);
      float rs = half_sum16(s0[r] + s1[r]);
      lrow[r] = lrow[r] * alpha + rs;
      acc0[r] *= alpha;
      acc1[r] *= alpha;
    }

    // ---- transpose P: C/D layout -> A layout via LDS ---------------------
    #pragma unroll
    for (int r = 0; r < 8; ++r) {
      int m = lg * 8 + r;
      p_lds[m * 32 + ln]      = (__bf16)s0[r];
      p_lds[m * 32 + 16 + ln] = (__bf16)s1[r];
    }
    asm volatile("s_wait_dscnt 0" ::: "memory");   // LDS RAW fence (same wave)
    v16bf pa = loadA((const __bf16*)&p_lds[ln * 32 + lg * 16]);

    // ---- out += P @ V, V^T stored [b,h,d][s] so B loads are contiguous ---
    v16bf vb0 = loadB(vT + ((size_t)((b * kH + h) * kD + ln))      * kS + base + lg * 16);
    v16bf vb1 = loadB(vT + ((size_t)((b * kH + h) * kD + 16 + ln)) * kS + base + lg * 16);
    acc0 = __builtin_amdgcn_wmma_f32_16x16x32_bf16(false, pa, false, vb0,
                                                   (short)0, acc0, false, false);
    acc1 = __builtin_amdgcn_wmma_f32_16x16x32_bf16(false, pa, false, vb1,
                                                   (short)0, acc1, false, false);
  }

  // ---- normalize and store (bf16, row-major, ready as next GEMM's A) ----
  #pragma unroll
  for (int r = 0; r < 8; ++r) {
    int row = qbase + lg * 8 + r;
    float inv = 1.0f / lrow[r];
    __bf16* orow = aout + ((size_t)(b * kS + row)) * kCM + h * kD;
    orow[ln]      = (__bf16)(acc0[r] * inv);
    orow[16 + ln] = (__bf16)(acc1[r] * inv);
  }
}

// ---------------------------------------------------------------------------
// Host: workspace layout + launch sequence
// ---------------------------------------------------------------------------
extern "C" void kernel_launch(void* const* d_in, const int* in_sizes, int n_in,
                              void* d_out, int out_size, void* d_ws, size_t ws_size,
                              hipStream_t stream) {
  const float* m_in  = (const float*)d_in[0];
  const float* z_in  = (const float*)d_in[1];
  const float* zmask = (const float*)d_in[2];
  const float* wnm   = (const float*)d_in[3];
  const float* wnz   = (const float*)d_in[4];
  const float* Wq    = (const float*)d_in[5];
  const float* Wk    = (const float*)d_in[6];
  const float* Wv    = (const float*)d_in[7];
  const float* Wz    = (const float*)d_in[8];
  const float* Wg    = (const float*)d_in[9];
  const float* bg    = (const float*)d_in[10];
  const float* Wo    = (const float*)d_in[11];
  const float* bo    = (const float*)d_in[12];
  float* out = (float*)d_out;
  char*  ws  = (char*)d_ws;

  const size_t NROW = (size_t)kB * kS;           // 32768
  const size_t WT_B = (size_t)kCM * kCM * 2;     // 128 KB per transposed weight

  __bf16* wqT    = (__bf16*)(ws + 0 * WT_B);
  __bf16* wkT    = (__bf16*)(ws + 1 * WT_B);
  __bf16* wvT    = (__bf16*)(ws + 2 * WT_B);
  __bf16* wgT    = (__bf16*)(ws + 3 * WT_B);
  __bf16* woT    = (__bf16*)(ws + 4 * WT_B);
  size_t off = 5 * WT_B;
  __bf16* m_norm = (__bf16*)(ws + off); off += NROW * kCM * 2;   // 16.8 MB
  __bf16* qb     = (__bf16*)(ws + off); off += NROW * kCM * 2;
  __bf16* kb     = (__bf16*)(ws + off); off += NROW * kCM * 2;
  __bf16* vTb    = (__bf16*)(ws + off); off += NROW * kCM * 2;
  float*  gbuf   = (float*) (ws + off); off += NROW * kCM * 4;   // 33.6 MB
  float*  biasb  = (float*) (ws + off); off += (size_t)kH * kS * kS * 4; // 8.4 MB
  __bf16* aoutb  = (__bf16*)(ws + off); off += NROW * kCM * 2;

  const float scale = 0.17677669529663687f;      // 1/sqrt(D)

  // 1. transpose+convert weights
  prep_w_kernel<<<256, 256, 0, stream>>>(Wq, wqT);
  prep_w_kernel<<<256, 256, 0, stream>>>(Wk, wkT);
  prep_w_kernel<<<256, 256, 0, stream>>>(Wv, wvT);
  prep_w_kernel<<<256, 256, 0, stream>>>(Wg, wgT);
  prep_w_kernel<<<256, 256, 0, stream>>>(Wo, woT);

  // 2. rmsnorm(m) -> bf16 (32768 rows, wave per row, 8 waves/block)
  rmsnorm_m_kernel<<<4096, 256, 0, stream>>>(m_in, wnm, m_norm);

  // 3. rmsnorm(z) + pair bias + mask (262144 pairs, wave per pair)
  pairbias_kernel<<<32768, 256, 0, stream>>>(z_in, zmask, wnz, Wz, biasb);

  // 4. projections: 32768 tiles per GEMM, 8 waves/block -> 4096 blocks
  gemm16_kernel<<<4096, 256, 0, stream>>>(m_norm, wqT, scale, nullptr, nullptr, qb,  nullptr, 0);
  gemm16_kernel<<<4096, 256, 0, stream>>>(m_norm, wkT, 1.0f,  nullptr, nullptr, kb,  nullptr, 0);
  gemm16_kernel<<<4096, 256, 0, stream>>>(m_norm, wvT, 1.0f,  nullptr, nullptr, vTb, nullptr, 2);
  gemm16_kernel<<<4096, 256, 0, stream>>>(m_norm, wgT, 1.0f,  bg,      nullptr, nullptr, gbuf, 3);

  // 5. flash attention: B*H*(S/16) = 16384 waves, one per block
  attn_kernel<<<kB * kH * (kS / 16), 32, 0, stream>>>(qb, kb, vTb, biasb, aoutb);

  // 6. output projection + bias + gate, straight into d_out
  gemm16_kernel<<<4096, 256, 0, stream>>>(aoutb, woT, 1.0f, bo, gbuf, nullptr, out, 4);

  (void)in_sizes; (void)n_in; (void)out_size; (void)ws_size;
}
